// EdgeFeatureExtractor_2894807957835
// MI455X (gfx1250) — compile-verified, matrix operand order checked by reading, not run on hardware
//
#include <hip/hip_runtime.h>

typedef __attribute__((ext_vector_type(16))) _Float16 v16h;
typedef __attribute__((ext_vector_type(8)))  float    v8f;

#define NUM_RBF 16
#define D_IN    19      // 16 RBF + 3 rel_norm
#define HIDDEN  64
#define CUTOFF  10.0f
#define NJ      32      // j-tile per loop iteration
#define APAD    36      // padded K row length for edge_input LDS (bank-friendly)
#define HPAD    68      // padded row length for hidden LDS

// Map WMMA 16-bit A/B fragment slot -> K index (CDNA5 ISA 7.12.2 layout).
// VGPR v in 0..7, each holds 2 f16; lanes 16..31 add +8 to K.
__device__ __forceinline__ int wmma_kbase(int v) {
    return (v < 4) ? (2 * v) : (16 + 2 * (v - 4));
}

__global__ __launch_bounds__(256)
void edge_feature_kernel(const float* __restrict__ coords,
                         const float* __restrict__ rbf_centers,
                         const float* __restrict__ rbf_width,
                         const float* __restrict__ W1,
                         const float* __restrict__ b1,
                         const float* __restrict__ W2,
                         const float* __restrict__ b2,
                         float* __restrict__ out_feat,
                         float* __restrict__ out_mask,
                         int B, int N)
{
    __shared__ _Float16 ldsA[NJ][APAD];   // edge_input tile, K padded to 32 (zeros)
    __shared__ _Float16 ldsH[NJ][HPAD];   // silu(hidden) tile, 32 x 64 used

    const int tid  = threadIdx.x;
    const int lane = tid & 31;
    const int wave = tid >> 5;

    const int bx = blockIdx.x;            // 0 .. B*N-1
    const int i  = bx % N;
    const int bb = bx / N;

    // wave -> output tile: jt selects j rows (0-15 / 16-31), nt selects 16 hidden cols
    const int jt = wave & 1;
    const int nt = wave >> 1;

    const int nlo  = lane & 15;           // N index within tile (B/C/D layouts)
    const int kgrp = (lane >> 4) * 8;     // +8 K offset for upper half-wave
    const int col  = nt * 16 + nlo;       // global hidden column for this lane

    // ---- Load weight fragments into registers once per block (f32 -> f16) ----
    v16h w1b;                             // W1 column tile as 32x16 B fragment (K>=19 -> 0)
    v16h w2b0, w2b1;                      // W2 column tile, K slices 0-31 / 32-63
    #pragma unroll
    for (int v = 0; v < 8; ++v) {
        const int kb = wmma_kbase(v);
        #pragma unroll
        for (int h = 0; h < 2; ++h) {
            const int k = kb + kgrp + h;
            w1b[2 * v + h]  = (_Float16)((k < D_IN) ? W1[k * HIDDEN + col] : 0.0f);
            w2b0[2 * v + h] = (_Float16)W2[(k)      * HIDDEN + col];
            w2b1[2 * v + h] = (_Float16)W2[(k + 32) * HIDDEN + col];
        }
    }
    const float bias1 = b1[col];
    const float bias2 = b2[col];

    const float width  = rbf_width[0];
    const float inv2w2 = __builtin_amdgcn_rcpf(2.0f * width * width);

    // center atom coordinates (uniform per block)
    const size_t ibase = ((size_t)bb * N + i) * 3;
    const float xi = coords[ibase + 0];
    const float yi = coords[ibase + 1];
    const float zi = coords[ibase + 2];

    // producer-phase mapping: 8 threads per j, each covers 4 K slots
    const int jl = tid & 31;              // j within tile
    const int g  = tid >> 5;              // k group: handles k = 4g..4g+3 (uniform per wave)

    const size_t rowI = (size_t)bb * N + i;

    for (int jb = 0; jb < N; jb += NJ) {
        __syncthreads();   // protect ldsA against previous iteration's readers

        // ---------- produce edge_input tile (32 j x 32 K, f16, zero padded) ----------
        {
            const int j = jb + jl;
            const size_t jbase = ((size_t)bb * N + j) * 3;
            const float rx = coords[jbase + 0] - xi;
            const float ry = coords[jbase + 1] - yi;
            const float rz = coords[jbase + 2] - zi;
            const float sq   = rx * rx + ry * ry + rz * rz;
            const float dist = (sq > 0.0f) ? __builtin_sqrtf(sq) : 0.0f;
            const float invd = __builtin_amdgcn_rcpf(dist + 1e-8f);
            #pragma unroll
            for (int kk = 0; kk < 4; ++kk) {
                const int k = g * 4 + kk;
                float val;
                if (k < NUM_RBF) {
                    const float d = dist - rbf_centers[k];
                    val = __expf(-(d * d) * inv2w2);
                } else if (k < D_IN) {
                    const float r = (k == 16) ? rx : ((k == 17) ? ry : rz);
                    val = r * invd;
                } else {
                    val = 0.0f;
                }
                ldsA[jl][k] = (_Float16)val;
            }
            if (g == 0) {
                out_mask[rowI * N + j] =
                    ((dist < CUTOFF) && (dist > 0.0f)) ? 1.0f : 0.0f;
            }
        }
        __syncthreads();

        // ---------- GEMM1: (16x32 edge_input) x (32x16 W1 tile) ----------
        const int row = jt * 16 + nlo;    // A-fragment row for this lane
        v16h a;
        #pragma unroll
        for (int v = 0; v < 8; ++v) {
            const int kb = wmma_kbase(v) + kgrp;
            a[2 * v]     = ldsA[row][kb];
            a[2 * v + 1] = ldsA[row][kb + 1];
        }
        v8f acc = {};
        acc = __builtin_amdgcn_wmma_f32_16x16x32_f16(
                  false, a, false, w1b, (short)0, acc, false, false);

        // SiLU on f32 accumulators, stage to LDS as f16 (D layout -> row major)
        #pragma unroll
        for (int r = 0; r < 8; ++r) {
            const float x = acc[r] + bias1;
            const float s = x * __builtin_amdgcn_rcpf(1.0f + __expf(-x));
            const int   m = r + ((lane >> 4) << 3);
            ldsH[jt * 16 + m][nt * 16 + nlo] = (_Float16)s;
        }
        __syncthreads();

        // ---------- GEMM2: (16x64 hidden) x (64x16 W2 tile), two K steps ----------
        v16h a0, a1;
        #pragma unroll
        for (int v = 0; v < 8; ++v) {
            const int kb = wmma_kbase(v) + kgrp;
            a0[2 * v]     = ldsH[row][kb];
            a0[2 * v + 1] = ldsH[row][kb + 1];
            a1[2 * v]     = ldsH[row][32 + kb];
            a1[2 * v + 1] = ldsH[row][32 + kb + 1];
        }
        v8f acc2 = {};
        acc2 = __builtin_amdgcn_wmma_f32_16x16x32_f16(
                   false, a0, false, w2b0, (short)0, acc2, false, false);
        acc2 = __builtin_amdgcn_wmma_f32_16x16x32_f16(
                   false, a1, false, w2b1, (short)0, acc2, false, false);

        // ---------- store D tile + bias2 (f32) ----------
        const size_t base = (rowI * N + (size_t)(jb + jt * 16)) * HIDDEN;
        #pragma unroll
        for (int r = 0; r < 8; ++r) {
            const int m = r + ((lane >> 4) << 3);
            out_feat[base + (size_t)m * HIDDEN + col] = acc2[r] + bias2;
        }
    }
}

extern "C" void kernel_launch(void* const* d_in, const int* in_sizes, int n_in,
                              void* d_out, int out_size, void* d_ws, size_t ws_size,
                              hipStream_t stream)
{
    (void)in_sizes; (void)n_in; (void)d_ws; (void)ws_size; (void)out_size;

    const float* coords      = (const float*)d_in[0];
    const float* rbf_centers = (const float*)d_in[1];
    const float* rbf_width   = (const float*)d_in[2];
    const float* W1          = (const float*)d_in[3];
    const float* b1          = (const float*)d_in[4];
    const float* W2          = (const float*)d_in[5];
    const float* b2          = (const float*)d_in[6];

    const int B = 4, N = 1024;

    float* out_feat = (float*)d_out;                              // (B,N,N,64)
    float* out_mask = out_feat + (size_t)B * N * N * HIDDEN;      // (B,N,N) as float

    dim3 grid(B * N), block(256);
    hipLaunchKernelGGL(edge_feature_kernel, grid, block, 0, stream,
                       coords, rbf_centers, rbf_width, W1, b1, W2, b2,
                       out_feat, out_mask, B, N);
}
